// CausalSelfAttention_22703197127379
// MI455X (gfx1250) — matrix operand order repaired and checked by reading, not compiled, not run on hardware
//
#include <hip/hip_runtime.h>

// ---------------- types ----------------
typedef __attribute__((ext_vector_type(16))) __bf16 v16bf;
typedef __attribute__((ext_vector_type(8)))  float  v8f;

#define EDIM 4096        // hidden size E
#define QSZ  4096        // NUM_Q_HEADS * HEAD_DIM
#define WQKV_LD 6144     // Q_SIZE + 2*KV_SIZE columns of w_qkv
#define BM 128
#define BN 128
#define BK 32

static __device__ __forceinline__ unsigned short f32_to_bf16(float f) {
    unsigned u = __float_as_uint(f);
    unsigned r = u + 0x7FFFu + ((u >> 16) & 1u);   // round-to-nearest-even
    return (unsigned short)(r >> 16);
}
static __device__ __forceinline__ float bf16_to_f32(unsigned short h) {
    return __uint_as_float(((unsigned)h) << 16);
}
// low 32 bits of a generic pointer to __shared__ = LDS byte offset (flat aperture rule)
static __device__ __forceinline__ unsigned lds_addr32(const void* p) {
    return (unsigned)(unsigned long long)p;
}
static __device__ __forceinline__ void async_ld_b128(unsigned lds, const void* g) {
    asm volatile("global_load_async_to_lds_b128 %0, %1, off"
                 :: "v"(lds), "v"(g) : "memory");
}

// ---------------- f32 -> bf16 (contiguous, vectorized) ----------------
__global__ __launch_bounds__(256) void cvt_f32_to_bf16(
    const float* __restrict__ in, unsigned short* __restrict__ out, long long n4) {
    long long i = (long long)blockIdx.x * blockDim.x + threadIdx.x;
    if (i >= n4) return;
    float4 v = ((const float4*)in)[i];
    uint2 p;
    p.x = ((unsigned)f32_to_bf16(v.y) << 16) | f32_to_bf16(v.x);
    p.y = ((unsigned)f32_to_bf16(v.w) << 16) | f32_to_bf16(v.z);
    ((uint2*)out)[i] = p;
}

// ---------------- f32 [K, ldw] -> bf16 [N, K] transpose-convert ----------------
__global__ __launch_bounds__(256) void cvt_transpose_bf16(
    const float* __restrict__ w, unsigned short* __restrict__ out,
    int K, int ldw, long long total) {
    long long i = (long long)blockIdx.x * blockDim.x + threadIdx.x;
    if (i >= total) return;
    int k = (int)(i % K);
    long long n = i / K;
    out[i] = f32_to_bf16(w[(size_t)k * ldw + n]);
}

// ---------------- NeoX RoPE on bf16 q buffer, in place ----------------
__global__ __launch_bounds__(256) void rope_neox_bf16(
    unsigned short* __restrict__ q, const int* __restrict__ positions, int T) {
    long long i = (long long)blockIdx.x * blockDim.x + threadIdx.x;
    if (i >= (long long)T * 32 * 64) return;
    int d = (int)(i & 63);
    int h = (int)((i >> 6) & 31);
    int t = (int)(i >> 11);
    float pos = (float)positions[t];
    float fr = pos * __expf(-((float)d * (1.0f / 64.0f)) * 9.210340371976184f);
    float c, s;
    __sincosf(fr, &s, &c);
    size_t base = (size_t)t * QSZ + (size_t)h * 128 + d;
    float x1 = bf16_to_f32(q[base]);
    float x2 = bf16_to_f32(q[base + 64]);
    q[base]      = f32_to_bf16(x1 * c - x2 * s);
    q[base + 64] = f32_to_bf16(x2 * c + x1 * s);
}

// ---------------- bf16 WMMA GEMM: C[M,N] = A[M,K] * Bt[N,K]^T ----------------
// Double-buffered LDS, async global->LDS DMA (ASYNCcnt), one barrier/iter.
// 256 threads = 8 wave32; waves in 4(M) x 2(N); wave tile 32x64 = 2x4 WMMA tiles.
template <bool OUT_BF16>
__global__ __launch_bounds__(256) void gemm_bf16_wmma(
    const unsigned short* __restrict__ A,
    const unsigned short* __restrict__ Bt,
    void* __restrict__ C, int M, int N, int K) {
    __shared__ unsigned short lA[2][BM * BK];   // [buf][row][k]
    __shared__ unsigned short lB[2][BN * BK];   // [buf][n][k] (N-major: contiguous K)

    const int tid  = threadIdx.x;
    const int lane = tid & 31;
    const int wave = tid >> 5;
    const int wm   = wave & 3;   // wave row  (4 along M)
    const int wn   = wave >> 2;  // wave col  (2 along N)
    const int bm   = blockIdx.x * BM;
    const int bn   = blockIdx.y * BN;
    const int lm   = lane & 15;
    const int lh   = lane >> 4;  // lane half

    // per-thread staging coordinates: 512 16B chunks per tile, 2 per thread
    const int row0 = tid >> 2;            // chunk 0: rows 0..63
    const int seg  = tid & 3;
    const int row1 = row0 + 64;           // chunk 1: rows 64..127

    unsigned ldsA0[2], ldsA1[2], ldsB0[2], ldsB1[2];
#pragma unroll
    for (int b = 0; b < 2; ++b) {
        ldsA0[b] = lds_addr32(&lA[b][row0 * BK + seg * 8]);
        ldsA1[b] = lds_addr32(&lA[b][row1 * BK + seg * 8]);
        ldsB0[b] = lds_addr32(&lB[b][row0 * BK + seg * 8]);
        ldsB1[b] = lds_addr32(&lB[b][row1 * BK + seg * 8]);
    }
    const unsigned short* gA0 = A  + (size_t)(bm + row0) * K + seg * 8;
    const unsigned short* gA1 = A  + (size_t)(bm + row1) * K + seg * 8;
    const unsigned short* gB0 = Bt + (size_t)(bn + row0) * K + seg * 8;
    const unsigned short* gB1 = Bt + (size_t)(bn + row1) * K + seg * 8;

    v8f acc[2][4];
#pragma unroll
    for (int mt = 0; mt < 2; ++mt)
#pragma unroll
        for (int nt = 0; nt < 4; ++nt)
#pragma unroll
            for (int r = 0; r < 8; ++r) acc[mt][nt][r] = 0.0f;

    // ---- prologue: DMA tile 0 into buffer 0 ----
    async_ld_b128(ldsA0[0], gA0);
    async_ld_b128(ldsA1[0], gA1);
    async_ld_b128(ldsB0[0], gB0);
    async_ld_b128(ldsB1[0], gB1);
    asm volatile("s_wait_asynccnt 0x0" ::: "memory");
    __syncthreads();

    for (int kt = 0; kt < K; kt += BK) {
        const int cur = (kt / BK) & 1;
        const int nxt = cur ^ 1;

        // ---- issue DMA for tile kt+BK into the other buffer (overlaps compute) ----
        if (kt + BK < K) {
            async_ld_b128(ldsA0[nxt], gA0 + kt + BK);
            async_ld_b128(ldsA1[nxt], gA1 + kt + BK);
            async_ld_b128(ldsB0[nxt], gB0 + kt + BK);
            async_ld_b128(ldsB1[nxt], gB1 + kt + BK);
            // pull the tile after that toward near cache (global_prefetch_b8)
            if (kt + 2 * BK < K) {
                __builtin_prefetch((const void*)(gA0 + kt + 2 * BK), 0, 3);
                __builtin_prefetch((const void*)(gB0 + kt + 2 * BK), 0, 3);
            }
        }

        const unsigned short* cA = lA[cur];
        const unsigned short* cB = lB[cur];

        // A fragments: 16x32 bf16 layout (ISA 7.12.2); contiguous per 16B half
        v16bf afrag[2];
#pragma unroll
        for (int mt = 0; mt < 2; ++mt) {
            int row = wm * 32 + mt * 16 + lm;
            union { v16bf v; unsigned u[8]; } fa;
#pragma unroll
            for (int j = 0; j < 8; ++j) {
                int k = ((j >> 2) << 4) + (lh << 3) + ((j & 3) << 1);
                fa.u[j] = *(const unsigned*)(cA + row * BK + k);
            }
            afrag[mt] = fa.v;
        }
        // B fragments: lane n = lane&15, laneHalf selects K 8h..8h+7 / 16+8h..16+8h+7
        v16bf bfrag[4];
#pragma unroll
        for (int nt = 0; nt < 4; ++nt) {
            int n = wn * 64 + nt * 16 + lm;
            union { v16bf v; uint4 q[2]; } fb;
            fb.q[0] = *(const uint4*)(cB + n * BK + lh * 8);
            fb.q[1] = *(const uint4*)(cB + n * BK + lh * 8 + 16);
            bfrag[nt] = fb.v;
        }

#pragma unroll
        for (int mt = 0; mt < 2; ++mt)
#pragma unroll
            for (int nt = 0; nt < 4; ++nt)
                acc[mt][nt] = __builtin_amdgcn_wmma_f32_16x16x32_bf16(
                    false, afrag[mt], false, bfrag[nt],
                    (short)0, acc[mt][nt], false, false);

        // my next-tile DMAs done; barrier makes all waves' DMAs + reads complete
        asm volatile("s_wait_asynccnt 0x0" ::: "memory");
        __syncthreads();
    }

    // store: C/D layout (ISA 7.12.2): row m = 8*laneHalf + r, col n = lane&15
#pragma unroll
    for (int mt = 0; mt < 2; ++mt)
#pragma unroll
        for (int nt = 0; nt < 4; ++nt)
#pragma unroll
            for (int r = 0; r < 8; ++r) {
                int m = bm + wm * 32 + mt * 16 + (lh << 3) + r;
                int n = bn + wn * 64 + nt * 16 + lm;
                if (OUT_BF16)
                    ((unsigned short*)C)[(size_t)m * N + n] = f32_to_bf16(acc[mt][nt][r]);
                else
                    ((float*)C)[(size_t)m * N + n] = acc[mt][nt][r];
            }
}

// ---------------- launch ----------------
extern "C" void kernel_launch(void* const* d_in, const int* in_sizes, int n_in,
                              void* d_out, int out_size, void* d_ws, size_t ws_size,
                              hipStream_t stream) {
    const float* hs    = (const float*)d_in[0];   // [T, 4096]
    const int*   pos   = (const int*)d_in[1];     // [T]
    const float* w_qkv = (const float*)d_in[2];   // [4096, 6144]
    const float* w_o   = (const float*)d_in[3];   // [4096, 4096]
    float*       out   = (float*)d_out;           // [T, 4096]

    const int T = in_sizes[0] / EDIM;             // 8192

    // workspace layout (bf16 buffers)
    unsigned short* hsb = (unsigned short*)d_ws;                         // T*4096
    unsigned short* wqt = hsb + (size_t)T * EDIM;                        // 4096*4096
    unsigned short* wot = wqt + (size_t)QSZ * EDIM;                      // 4096*4096
    unsigned short* qbf = wot + (size_t)EDIM * QSZ;                      // T*4096

    // 1) hidden f32 -> bf16
    {
        long long n4 = ((long long)T * EDIM) / 4;
        cvt_f32_to_bf16<<<(unsigned)((n4 + 255) / 256), 256, 0, stream>>>(hs, hsb, n4);
    }
    // 2) w_qkv Q-slice [E, first 4096 of 6144] -> bf16 [N=4096, K=4096]
    {
        long long total = (long long)QSZ * EDIM;
        cvt_transpose_bf16<<<(unsigned)((total + 255) / 256), 256, 0, stream>>>(
            w_qkv, wqt, EDIM, WQKV_LD, total);
    }
    // 3) w_o [4096,4096] -> bf16 transposed
    {
        long long total = (long long)EDIM * QSZ;
        cvt_transpose_bf16<<<(unsigned)((total + 255) / 256), 256, 0, stream>>>(
            w_o, wot, QSZ, EDIM, total);
    }
    // 4) q = hidden @ Wq   (bf16 out)
    {
        dim3 grid(T / BM, QSZ / BN);
        gemm_bf16_wmma<true><<<grid, 256, 0, stream>>>(hsb, wqt, qbf, T, QSZ, EDIM);
    }
    // 5) RoPE (NeoX) in place on q
    {
        long long total = (long long)T * 32 * 64;
        rope_neox_bf16<<<(unsigned)((total + 255) / 256), 256, 0, stream>>>(qbf, pos, T);
    }
    // 6) out = q @ Wo   (f32 out)
    {
        dim3 grid(T / BM, EDIM / BN);
        gemm_bf16_wmma<false><<<grid, 256, 0, stream>>>(qbf, wot, out, T, EDIM, QSZ);
    }
}
